// GATBlock_6871947674031
// MI455X (gfx1250) — matrix operand order repaired and checked by reading, not compiled, not run on hardware
//
#include <hip/hip_runtime.h>

// ---------------------------------------------------------------------------
// GAT 2-layer block for MI455X (gfx1250, wave32, WMMA).
// ---------------------------------------------------------------------------

typedef __attribute__((ext_vector_type(16))) _Float16 v16h;
typedef __attribute__((ext_vector_type(8)))  _Float16 v8h;
typedef __attribute__((ext_vector_type(8)))  float    v8f;

#define NNODES 20000
#define NEDGES 320000
#define FIN    128
#define NCOL   256      // H*HID == H*OUT
#define NHEAD  8
#define DHEAD  32
#define NEG_SLOPE 0.2f

// ---- monotonic float <-> uint mapping for atomicMax-based segment max ------
__device__ __forceinline__ unsigned f2mono(float f) {
    unsigned u = __float_as_uint(f);
    return (u & 0x80000000u) ? ~u : (u | 0x80000000u);
}
__device__ __forceinline__ float mono2f(unsigned u) {
    unsigned b = (u & 0x80000000u) ? (u & 0x7fffffffu) : ~u;
    return __uint_as_float(b);
}

// ---------------------------------------------------------------------------
// fp32 -> fp16 convert
__global__ void cvt_f16_kernel(const float* __restrict__ x,
                               _Float16* __restrict__ y, int n) {
    int i = blockIdx.x * blockDim.x + threadIdx.x;
    if (i < n) y[i] = (_Float16)x[i];
}

// W [K, NCOLS] row-major fp32  ->  Wt [NCOLS, K] row-major fp16 (B col-major)
__global__ void transpose_cvt_kernel(const float* __restrict__ W,
                                     _Float16* __restrict__ Wt,
                                     int K, int Ncols) {
    int i = blockIdx.x * blockDim.x + threadIdx.x;
    if (i >= K * Ncols) return;
    int k = i / Ncols, n = i - k * Ncols;
    Wt[(size_t)n * K + k] = (_Float16)W[i];
}

// ---------------------------------------------------------------------------
// WMMA GEMM:  C[M, 256] = A[M, K] * B[K, 256],  A fp16 row-major,
// B supplied transposed as Bt[256, K] fp16.  One block = 16-row stripe,
// 4 waves, each wave emits four 16x16 tiles (reusing its A fragments).
template <int K>
__global__ void gemm_wmma_kernel(const _Float16* __restrict__ A,
                                 const _Float16* __restrict__ Bt,
                                 float* __restrict__ C, int M) {
    constexpr int KSTEPS = K / 32;
    const int lane = threadIdx.x & 31;
    const int wave = threadIdx.x >> 5;
    const int m0 = blockIdx.x * 16;
    if (m0 + 16 > M) return;                 // uniform per block -> EXEC stays all-1s

    const int r    = lane & 15;
    const int hiA  = (lane >> 4) ? 8  : 0;   // A: lanes>=16 hold K+8 .. / K+24..
    const int hiB  = (lane >> 4) ? 8  : 0;   // B: lanes>=16 hold K+8 .. / K+24..

    // Preload all A fragments for this 16-row stripe (ISA 16-bit A 16x32 layout).
    v16h a[KSTEPS];
    const _Float16* arow = A + (size_t)(m0 + r) * K;
    #pragma unroll
    for (int ks = 0; ks < KSTEPS; ++ks) {
        const int kk = ks * 32;
        v8h lo = *(const v8h*)(arow + kk + hiA);        // K kk+hiA   .. +7
        v8h hi = *(const v8h*)(arow + kk + 16 + hiA);   // K kk+16+hiA.. +7
        #pragma unroll
        for (int i = 0; i < 8; ++i) { a[ks][i] = lo[i]; a[ks][i + 8] = hi[i]; }
    }

    #pragma unroll
    for (int j = 0; j < 4; ++j) {
        const int n0 = (wave * 4 + j) * 16;
        const _Float16* bcol = Bt + (size_t)(n0 + r) * K + hiB;
        v8f c = {};
        #pragma unroll
        for (int ks = 0; ks < KSTEPS; ++ks) {
            v16h b;
            v8h lo = *(const v8h*)(bcol + ks * 32);      // K hiB .. +7
            v8h hi = *(const v8h*)(bcol + ks * 32 + 16); // K 16+hiB .. +7
            #pragma unroll
            for (int i = 0; i < 8; ++i) { b[i] = lo[i]; b[i + 8] = hi[i]; }
            c = __builtin_amdgcn_wmma_f32_16x16x32_f16(
                    false, a[ks], false, b, (short)0, c, false, false);
        }
        // C/D layout: VGPR i -> row m0 + i (+8 for lanes 16..31), col n0 + (lane&15)
        const int rbase = m0 + ((lane >> 4) ? 8 : 0);
        const int col   = n0 + r;
        #pragma unroll
        for (int i = 0; i < 8; ++i)
            C[(size_t)(rbase + i) * NCOL + col] = c[i];
    }
}

// ---------------------------------------------------------------------------
// el[n,h] = <feat[n,h,:], al[h,:]>,  er likewise.
__global__ void node_scores_kernel(const float* __restrict__ feat,
                                   const float* __restrict__ al,
                                   const float* __restrict__ ar,
                                   float* __restrict__ el,
                                   float* __restrict__ er) {
    int i = blockIdx.x * blockDim.x + threadIdx.x;     // n*NHEAD + h
    if (i >= NNODES * NHEAD) return;
    int n = i >> 3, h = i & 7;
    const float* f = feat + (size_t)n * NCOL + h * DHEAD;
    const float* a = al + h * DHEAD;
    const float* b = ar + h * DHEAD;
    float sl = 0.f, sr = 0.f;
    #pragma unroll
    for (int d = 0; d < DHEAD; ++d) { float v = f[d]; sl += v * a[d]; sr += v * b[d]; }
    el[i] = sl; er[i] = sr;
}

// e = leaky_relu(el[src]+er[dst]); segment-max into emax via monotonic atomicMax.
__global__ void edge_scores_kernel(const int* __restrict__ src,
                                   const int* __restrict__ dst,
                                   const float* __restrict__ el,
                                   const float* __restrict__ er,
                                   float* __restrict__ ebuf,
                                   unsigned* __restrict__ emax) {
    int i = blockIdx.x * blockDim.x + threadIdx.x;     // e*NHEAD + h
    if (i >= NEDGES * NHEAD) return;
    int e = i >> 3, h = i & 7;
    int s = src[e], d = dst[e];
    float v = el[s * NHEAD + h] + er[d * NHEAD + h];
    v = v > 0.f ? v : NEG_SLOPE * v;
    ebuf[i] = v;
    atomicMax(&emax[d * NHEAD + h], f2mono(v));
}

// ee = exp(e - emax[dst]); denom[dst] += ee  (stable edge softmax).
__global__ void edge_exp_kernel(const int* __restrict__ dst,
                                float* __restrict__ ebuf,
                                const unsigned* __restrict__ emax,
                                float* __restrict__ den) {
    int i = blockIdx.x * blockDim.x + threadIdx.x;
    if (i >= NEDGES * NHEAD) return;
    int e = i >> 3, h = i & 7;
    int d = dst[e];
    float m = mono2f(emax[d * NHEAD + h]);
    float x = __expf(ebuf[i] - m);
    ebuf[i] = x;
    atomicAdd(&den[d * NHEAD + h], x);
}

// One wave per edge: lane owns one d-column across the 8 heads.
//   acc[dst, h, d] += feat[src, h, d] * alpha[e, h]
// Each lane computes alpha for head (lane&7) once (coalesced loads of the
// 8 consecutive ee/denom values), then the per-head loop broadcasts it with
// a wave32 shuffle (lane-permute HW, no LDS memory) instead of re-loading
// broadcast addresses 8x per lane.
__global__ void edge_aggregate_kernel(const int* __restrict__ src,
                                      const int* __restrict__ dst,
                                      const float* __restrict__ ebuf,
                                      const float* __restrict__ den,
                                      const float* __restrict__ feat,
                                      float* __restrict__ acc) {
    int e = blockIdx.x * (blockDim.x >> 5) + (threadIdx.x >> 5);
    if (e >= NEDGES) return;
    const int lane = threadIdx.x & 31;
    const int s = src[e], d = dst[e];
    const float* fs = feat + (size_t)s * NCOL;
    float*       ad = acc  + (size_t)d * NCOL;
    __builtin_prefetch(fs + lane, 0, 0);               // global_prefetch_b8

    const int h0 = lane & 7;                            // my head
    const float my_alpha = ebuf[e * NHEAD + h0] / den[d * NHEAD + h0];

    #pragma unroll
    for (int h = 0; h < NHEAD; ++h) {
        float alpha = __shfl(my_alpha, h, 32);          // broadcast lane h's alpha
        int col = h * DHEAD + lane;
        atomicAdd(&ad[col], fs[col] * alpha);
    }
}

// h[n,d] = mean_h relu(acc[n,h,d] + b[h,d])
__global__ void relu_mean_kernel(const float* __restrict__ acc,
                                 const float* __restrict__ bias,
                                 float* __restrict__ out) {
    int i = blockIdx.x * blockDim.x + threadIdx.x;     // n*DHEAD + d
    if (i >= NNODES * DHEAD) return;
    int n = i >> 5, d = i & 31;
    float s = 0.f;
    #pragma unroll
    for (int h = 0; h < NHEAD; ++h) {
        float v = acc[(size_t)n * NCOL + h * DHEAD + d] + bias[h * DHEAD + d];
        s += v > 0.f ? v : 0.f;
    }
    out[i] = s * (1.f / NHEAD);
}

// out[n,d] = mean_h (acc[n,h,d] + b[h,d])
__global__ void mean_kernel(const float* __restrict__ acc,
                            const float* __restrict__ bias,
                            float* __restrict__ out) {
    int i = blockIdx.x * blockDim.x + threadIdx.x;
    if (i >= NNODES * DHEAD) return;
    int n = i >> 5, d = i & 31;
    float s = 0.f;
    #pragma unroll
    for (int h = 0; h < NHEAD; ++h)
        s += acc[(size_t)n * NCOL + h * DHEAD + d] + bias[h * DHEAD + d];
    out[i] = s * (1.f / NHEAD);
}

// ---------------------------------------------------------------------------
extern "C" void kernel_launch(void* const* d_in, const int* in_sizes, int n_in,
                              void* d_out, int out_size, void* d_ws, size_t ws_size,
                              hipStream_t stream) {
    const float* x   = (const float*)d_in[0];
    const int*   src = (const int*)  d_in[1];
    const int*   dst = (const int*)  d_in[2];
    const float* W1  = (const float*)d_in[3];
    const float* al1 = (const float*)d_in[4];
    const float* ar1 = (const float*)d_in[5];
    const float* b1  = (const float*)d_in[6];
    const float* W2  = (const float*)d_in[7];
    const float* al2 = (const float*)d_in[8];
    const float* ar2 = (const float*)d_in[9];
    const float* b2  = (const float*)d_in[10];
    float* out = (float*)d_out;

    // ---- workspace carve-up (256B aligned) ----
    char* ws = (char*)d_ws;
    size_t off = 0;
    auto carve = [&](size_t bytes) -> char* {
        char* p = ws + off;
        off = (off + bytes + 255) & ~(size_t)255;
        return p;
    };
    _Float16* xh    = (_Float16*)carve((size_t)NNODES * FIN * 2);
    _Float16* w1t   = (_Float16*)carve((size_t)NCOL * FIN * 2);
    _Float16* w2t   = (_Float16*)carve((size_t)NCOL * DHEAD * 2);
    _Float16* hh    = (_Float16*)carve((size_t)NNODES * DHEAD * 2);
    float*    feat  = (float*)   carve((size_t)NNODES * NCOL * 4);
    float*    acc   = (float*)   carve((size_t)NNODES * NCOL * 4);
    float*    el    = (float*)   carve((size_t)NNODES * NHEAD * 4);
    float*    er    = (float*)   carve((size_t)NNODES * NHEAD * 4);
    unsigned* emax  = (unsigned*)carve((size_t)NNODES * NHEAD * 4);
    float*    den   = (float*)   carve((size_t)NNODES * NHEAD * 4);
    float*    ebuf  = (float*)   carve((size_t)NEDGES * NHEAD * 4);
    float*    hbuf  = (float*)   carve((size_t)NNODES * DHEAD * 4);
    (void)ws_size; (void)in_sizes; (void)n_in; (void)out_size;

    const int TPB = 256;
    const int gridNodeHead = (NNODES * NHEAD + TPB - 1) / TPB;
    const int gridEdgeHead = (NEDGES * NHEAD + TPB - 1) / TPB;
    const int gridNodeD    = (NNODES * DHEAD + TPB - 1) / TPB;
    const int gridEdgeWave = (NEDGES + (TPB / 32) - 1) / (TPB / 32);

    // ---- precision staging ----
    cvt_f16_kernel<<<(NNODES * FIN + TPB - 1) / TPB, TPB, 0, stream>>>(x, xh, NNODES * FIN);
    transpose_cvt_kernel<<<(FIN * NCOL + TPB - 1) / TPB, TPB, 0, stream>>>(W1, w1t, FIN, NCOL);
    transpose_cvt_kernel<<<(DHEAD * NCOL + TPB - 1) / TPB, TPB, 0, stream>>>(W2, w2t, DHEAD, NCOL);

    // =========================== layer 1 ===========================
    gemm_wmma_kernel<FIN><<<NNODES / 16, 128, 0, stream>>>(xh, w1t, feat, NNODES);
    node_scores_kernel<<<gridNodeHead, TPB, 0, stream>>>(feat, al1, ar1, el, er);

    hipMemsetAsync(emax, 0, (size_t)NNODES * NHEAD * 4, stream);  // mono(0) < mono(any finite)
    hipMemsetAsync(den,  0, (size_t)NNODES * NHEAD * 4, stream);
    hipMemsetAsync(acc,  0, (size_t)NNODES * NCOL  * 4, stream);

    edge_scores_kernel<<<gridEdgeHead, TPB, 0, stream>>>(src, dst, el, er, ebuf, emax);
    edge_exp_kernel   <<<gridEdgeHead, TPB, 0, stream>>>(dst, ebuf, emax, den);
    edge_aggregate_kernel<<<gridEdgeWave, TPB, 0, stream>>>(src, dst, ebuf, den, feat, acc);
    relu_mean_kernel<<<gridNodeD, TPB, 0, stream>>>(acc, b1, hbuf);

    // =========================== layer 2 ===========================
    cvt_f16_kernel<<<(NNODES * DHEAD + TPB - 1) / TPB, TPB, 0, stream>>>(hbuf, hh, NNODES * DHEAD);
    gemm_wmma_kernel<DHEAD><<<NNODES / 16, 128, 0, stream>>>(hh, w2t, feat, NNODES);
    node_scores_kernel<<<gridNodeHead, TPB, 0, stream>>>(feat, al2, ar2, el, er);

    hipMemsetAsync(emax, 0, (size_t)NNODES * NHEAD * 4, stream);
    hipMemsetAsync(den,  0, (size_t)NNODES * NHEAD * 4, stream);
    hipMemsetAsync(acc,  0, (size_t)NNODES * NCOL  * 4, stream);

    edge_scores_kernel<<<gridEdgeHead, TPB, 0, stream>>>(src, dst, el, er, ebuf, emax);
    edge_exp_kernel   <<<gridEdgeHead, TPB, 0, stream>>>(dst, ebuf, emax, den);
    edge_aggregate_kernel<<<gridEdgeWave, TPB, 0, stream>>>(src, dst, ebuf, den, feat, acc);
    mean_kernel<<<gridNodeD, TPB, 0, stream>>>(acc, b2, out);
}
